// Encoder_61924838474399
// MI455X (gfx1250) — matrix operand order repaired and checked by reading, not compiled, not run on hardware
//
#include <hip/hip_runtime.h>

#define HIDDEN 1024
#define HEADS 16
#define HEAD 64
#define NB 2
#define LSEQ 2048
#define MROWS (NB * LSEQ) /* 4096 */

typedef __attribute__((ext_vector_type(16))) __bf16 v16bf;
typedef __attribute__((ext_vector_type(8)))  float  v8f;
typedef __attribute__((ext_vector_type(8)))  unsigned short u8s;

union Tile16 { v16bf v; u8s h[2]; unsigned short s[16]; };

// fp32 -> bf16 bits, round-to-nearest-even (bit twiddle; no __bf16 arithmetic needed)
__device__ __forceinline__ unsigned short f2bf(float f) {
    unsigned u = __float_as_uint(f);
    unsigned r = u + 0x7FFFu + ((u >> 16) & 1u);
    return (unsigned short)(r >> 16);
}

__device__ __forceinline__ v8f wmma_bf16(v16bf a, v16bf b, v8f c) {
    // v_wmma_f32_16x16x32_bf16: D(16x16 f32) = A(16x32 bf16) x B(32x16 bf16) + C
    return __builtin_amdgcn_wmma_f32_16x16x32_bf16(false, a, false, b, (short)0, c,
                                                   false, false);
}

// ds_swizzle group-of-32 xor butterfly: offset[14:10]=xor, [9:5]=or=0, [4:0]=and=0x1F.
// Immediate-encoded (no index VGPRs, unlike ds_bpermute), one DS op per step.
template <int XM>
__device__ __forceinline__ float swz_xor(float v) {
    return __int_as_float(
        __builtin_amdgcn_ds_swizzle(__float_as_int(v), (XM << 10) | 0x1F));
}

// A-matrix fragment (16x32 bf16): lane holds row M=lane&15; lane-half selects
// K=0..15 (lo) / K=16..31 (hi), contiguous. rowk0 points at K=0 of the
// 32-wide K window for this lane's row; 16B-aligned chunks.
__device__ __forceinline__ v16bf load_a32(const unsigned short* rowk0, int hi) {
    Tile16 t;
    t.h[0] = *(const u8s*)(rowk0 + 16 * hi);
    t.h[1] = *(const u8s*)(rowk0 + 16 * hi + 8);
    return t.v;
}

// B-matrix fragment (32x16 bf16): lane holds column N=lane&15; lanes 0-15 carry
// K=0..15, lanes 16-31 carry K=16..31 (16 contiguous bf16 along K).
__device__ __forceinline__ v16bf load_b32(const unsigned short* colk0, int hi) {
    Tile16 t;
    const unsigned short* p = colk0 + 16 * hi;
    t.h[0] = *(const u8s*)(p);
    t.h[1] = *(const u8s*)(p + 8);
    return t.v;
}

__global__ void cvt_f32_bf16(const float* __restrict__ in,
                             unsigned short* __restrict__ out, int n) {
    int i = blockIdx.x * blockDim.x + threadIdx.x;
    if (i < n) out[i] = f2bf(in[i]);
}

// C[M x 1024] = A[M x 1024] * W^T + bias,  W is [1024 x 1024] row-major (torch Linear).
// Each wave: one 16-row x 64-col strip; K loop in steps of 32; 4 WMMAs per step.
template <bool BF16OUT>
__global__ void gemm_xwt(const unsigned short* __restrict__ A,
                         const unsigned short* __restrict__ W,
                         const float* __restrict__ bias,
                         float* __restrict__ outF,
                         unsigned short* __restrict__ outB) {
    const int lane = threadIdx.x & 31;
    const int hi = lane >> 4, ln = lane & 15;
    const int wid = blockIdx.x * (blockDim.x >> 5) + (threadIdx.x >> 5);
    const int mtile = wid >> 4;          // 256 m-tiles
    const int nbase = (wid & 15) * 64;   // 16 n-strips of 64

    v8f acc0 = {}, acc1 = {}, acc2 = {}, acc3 = {};
    const unsigned short* arow = A + (size_t)(mtile * 16 + ln) * HIDDEN;
    const unsigned short* b0 = W + (size_t)(nbase + 0 + ln) * HIDDEN;
    const unsigned short* b1 = W + (size_t)(nbase + 16 + ln) * HIDDEN;
    const unsigned short* b2 = W + (size_t)(nbase + 32 + ln) * HIDDEN;
    const unsigned short* b3 = W + (size_t)(nbase + 48 + ln) * HIDDEN;

#pragma unroll 4
    for (int k0 = 0; k0 < HIDDEN; k0 += 32) {
        __builtin_prefetch(arow + k0 + 64, 0, 1);
        __builtin_prefetch(b0 + k0 + 64, 0, 1);
        v16bf a = load_a32(arow + k0, hi);
        acc0 = wmma_bf16(a, load_b32(b0 + k0, hi), acc0);
        acc1 = wmma_bf16(a, load_b32(b1 + k0, hi), acc1);
        acc2 = wmma_bf16(a, load_b32(b2 + k0, hi), acc2);
        acc3 = wmma_bf16(a, load_b32(b3 + k0, hi), acc3);
    }

    const float bv0 = bias[nbase + 0 + ln];
    const float bv1 = bias[nbase + 16 + ln];
    const float bv2 = bias[nbase + 32 + ln];
    const float bv3 = bias[nbase + 48 + ln];

    // C/D layout: VGPR r holds row (r + 8*hi), column = ln.
#pragma unroll
    for (int r = 0; r < 8; ++r) {
        const int row = mtile * 16 + r + 8 * hi;
        const size_t o = (size_t)row * HIDDEN + nbase + ln;
        float c0 = acc0[r] + bv0, c1 = acc1[r] + bv1;
        float c2 = acc2[r] + bv2, c3 = acc3[r] + bv3;
        if (BF16OUT) {
            outB[o] = f2bf(c0); outB[o + 16] = f2bf(c1);
            outB[o + 32] = f2bf(c2); outB[o + 48] = f2bf(c3);
        } else {
            outF[o] = c0; outF[o + 16] = c1;
            outF[o + 32] = c2; outF[o + 48] = c3;
        }
    }
}

// Fused flash attention. One wave per (batch, head, 16-row q-tile). No 1/sqrt(d) scale
// (matches reference). Keys processed 32 at a time; S and P never leave the WGP.
__global__ void flash_attn(const unsigned short* __restrict__ Q,
                           const unsigned short* __restrict__ K,
                           const unsigned short* __restrict__ V,
                           unsigned short* __restrict__ CTX) {
    __shared__ __align__(16) unsigned short pbuf[4][16 * 32];  // P tile per wave
    const int lane = threadIdx.x & 31;
    const int hi = lane >> 4, ln = lane & 15;
    const int w = threadIdx.x >> 5;
    const int wid = blockIdx.x * 4 + w;
    const int qt = wid & 127;            // 128 q-tiles
    const int h = (wid >> 7) & 15;       // 16 heads
    const int b = wid >> 11;             // 2 batches

    const size_t qoff = ((size_t)(b * LSEQ) + qt * 16 + ln) * HIDDEN + h * HEAD;
    const v16bf qa0 = load_a32(Q + qoff, hi);       // head dims 0..31
    const v16bf qa1 = load_a32(Q + qoff + 32, hi);  // head dims 32..63

    float rowmax[8], rowsum[8];
    v8f acc0 = {}, acc1 = {}, acc2 = {}, acc3 = {};
#pragma unroll
    for (int r = 0; r < 8; ++r) { rowmax[r] = -1e30f; rowsum[r] = 0.f; }

    for (int kb = 0; kb < LSEQ; kb += 32) {
        // S = Q(16x64) * K^T : two 16x16 tiles (keys kb..+15, kb+16..+31),
        // each accumulating over head dim via two K=32 WMMAs.
        const unsigned short* kp0 =
            K + ((size_t)(b * LSEQ) + kb + ln) * HIDDEN + h * HEAD;
        const unsigned short* kp1 = kp0 + (size_t)16 * HIDDEN;
        v8f s0 = {}, s1 = {};
        s0 = wmma_bf16(qa0, load_b32(kp0, hi), s0);
        s0 = wmma_bf16(qa1, load_b32(kp0 + 32, hi), s0);
        s1 = wmma_bf16(qa0, load_b32(kp1, hi), s1);
        s1 = wmma_bf16(qa1, load_b32(kp1 + 32, hi), s1);

        // Row max: 8 independent butterflies per step so DS ops cluster per step
        // (row = r + 8*hi; xor masks <=8 stay inside each 16-lane half).
        float mx[8];
#pragma unroll
        for (int r = 0; r < 8; ++r) mx[r] = fmaxf(s0[r], s1[r]);
#pragma unroll
        for (int r = 0; r < 8; ++r) mx[r] = fmaxf(mx[r], swz_xor<1>(mx[r]));
#pragma unroll
        for (int r = 0; r < 8; ++r) mx[r] = fmaxf(mx[r], swz_xor<2>(mx[r]));
#pragma unroll
        for (int r = 0; r < 8; ++r) mx[r] = fmaxf(mx[r], swz_xor<4>(mx[r]));
#pragma unroll
        for (int r = 0; r < 8; ++r) mx[r] = fmaxf(mx[r], swz_xor<8>(mx[r]));

        float al[8], ps[8];
#pragma unroll
        for (int r = 0; r < 8; ++r) {
            const float mnew = fmaxf(rowmax[r], mx[r]);
            al[r] = __expf(rowmax[r] - mnew);
            rowmax[r] = mnew;
            const float p0 = __expf(s0[r] - mnew);
            const float p1 = __expf(s1[r] - mnew);
            ps[r] = p0 + p1;
            const int rr = (r + 8 * hi) * 32;  // C-layout row -> LDS row
            pbuf[w][rr + ln] = f2bf(p0);
            pbuf[w][rr + 16 + ln] = f2bf(p1);
        }

        // Row sum butterflies (same clustering).
#pragma unroll
        for (int r = 0; r < 8; ++r) ps[r] += swz_xor<1>(ps[r]);
#pragma unroll
        for (int r = 0; r < 8; ++r) ps[r] += swz_xor<2>(ps[r]);
#pragma unroll
        for (int r = 0; r < 8; ++r) ps[r] += swz_xor<4>(ps[r]);
#pragma unroll
        for (int r = 0; r < 8; ++r) ps[r] += swz_xor<8>(ps[r]);

#pragma unroll
        for (int r = 0; r < 8; ++r) {
            rowsum[r] = rowsum[r] * al[r] + ps[r];
            acc0[r] *= al[r]; acc1[r] *= al[r];
            acc2[r] *= al[r]; acc3[r] *= al[r];
        }

        // DS ops are wave-in-order; wait ensures LDS stores landed before re-read.
        asm volatile("s_wait_dscnt 0" ::: "memory");

        // P re-read in A-layout (16x32), V fragments in B-layout (strided gather).
        const v16bf pa = load_a32(&pbuf[w][ln * 32], hi);
        const unsigned short* vbase =
            V + ((size_t)(b * LSEQ) + kb + 16 * hi) * HIDDEN + h * HEAD + ln;
        Tile16 v0t, v1t, v2t, v3t;
#pragma unroll
        for (int j = 0; j < 16; ++j) {
            const unsigned short* vr = vbase + (size_t)j * HIDDEN;
            v0t.s[j] = vr[0];  v1t.s[j] = vr[16];
            v2t.s[j] = vr[32]; v3t.s[j] = vr[48];
        }
        acc0 = wmma_bf16(pa, v0t.v, acc0);
        acc1 = wmma_bf16(pa, v1t.v, acc1);
        acc2 = wmma_bf16(pa, v2t.v, acc2);
        acc3 = wmma_bf16(pa, v3t.v, acc3);
        asm volatile("" ::: "memory");  // keep next iter's LDS stores after these reads
    }

#pragma unroll
    for (int r = 0; r < 8; ++r) {
        const float inv = 1.0f / rowsum[r];
        const size_t o =
            ((size_t)(b * LSEQ) + qt * 16 + r + 8 * hi) * HIDDEN + h * HEAD + ln;
        CTX[o] = f2bf(acc0[r] * inv);
        CTX[o + 16] = f2bf(acc1[r] * inv);
        CTX[o + 32] = f2bf(acc2[r] * inv);
        CTX[o + 48] = f2bf(acc3[r] * inv);
    }
}

extern "C" void kernel_launch(void* const* d_in, const int* in_sizes, int n_in,
                              void* d_out, int out_size, void* d_ws, size_t ws_size,
                              hipStream_t stream) {
    (void)in_sizes; (void)n_in; (void)out_size; (void)ws_size;
    const float* x  = (const float*)d_in[0];
    const float* Wq = (const float*)d_in[1];
    const float* bq = (const float*)d_in[2];
    const float* Wk = (const float*)d_in[3];
    const float* bk = (const float*)d_in[4];
    const float* Wv = (const float*)d_in[5];
    const float* bv = (const float*)d_in[6];
    const float* Wo = (const float*)d_in[7];
    const float* bo = (const float*)d_in[8];

    char* ws = (char*)d_ws;
    const size_t MB = 1u << 20;
    unsigned short* xb   = (unsigned short*)(ws + 0 * MB);   // 8 MB
    unsigned short* wqb  = (unsigned short*)(ws + 8 * MB);   // 2 MB
    unsigned short* wkb  = (unsigned short*)(ws + 10 * MB);  // 2 MB
    unsigned short* wvb  = (unsigned short*)(ws + 12 * MB);  // 2 MB
    unsigned short* wob  = (unsigned short*)(ws + 14 * MB);  // 2 MB
    unsigned short* qb   = (unsigned short*)(ws + 16 * MB);  // 8 MB
    unsigned short* kbuf = (unsigned short*)(ws + 24 * MB);  // 8 MB
    unsigned short* vbuf = (unsigned short*)(ws + 32 * MB);  // 8 MB
    unsigned short* ctxb = (unsigned short*)(ws + 40 * MB);  // 8 MB

    const int nx = MROWS * HIDDEN;      // 4,194,304
    const int nw = HIDDEN * HIDDEN;     // 1,048,576
    cvt_f32_bf16<<<(nx + 255) / 256, 256, 0, stream>>>(x, xb, nx);
    cvt_f32_bf16<<<(nw + 255) / 256, 256, 0, stream>>>(Wq, wqb, nw);
    cvt_f32_bf16<<<(nw + 255) / 256, 256, 0, stream>>>(Wk, wkb, nw);
    cvt_f32_bf16<<<(nw + 255) / 256, 256, 0, stream>>>(Wv, wvb, nw);
    cvt_f32_bf16<<<(nw + 255) / 256, 256, 0, stream>>>(Wo, wob, nw);

    // (4096/16 m-tiles) * (1024/64 n-strips) = 4096 waves -> 512 blocks x 8 waves
    gemm_xwt<true><<<512, 256, 0, stream>>>(xb, wqb, bq, nullptr, qb);
    gemm_xwt<true><<<512, 256, 0, stream>>>(xb, wkb, bk, nullptr, kbuf);
    gemm_xwt<true><<<512, 256, 0, stream>>>(xb, wvb, bv, nullptr, vbuf);

    // 2 batches * 16 heads * 128 q-tiles = 4096 waves -> 1024 blocks x 4 waves
    flash_attn<<<1024, 128, 0, stream>>>(qb, kbuf, vbuf, ctxb);

    gemm_xwt<false><<<512, 256, 0, stream>>>(ctxb, wob, bo, (float*)d_out, nullptr);
}
